// PreRoutGNN_63660005261509
// MI455X (gfx1250) — compile-verified, matrix operand order checked by reading, not run on hardware
//
#include <hip/hip_runtime.h>
#include <math.h>

#define N_NODES  400000
#define E_NET_N  400000
#define E_CELL_N 600000

static_assert(N_NODES  % 64 == 0, "full tiles");
static_assert(E_NET_N  % 64 == 0, "full tiles");
static_assert(E_CELL_N % 64 == 0, "full tiles");

typedef __attribute__((ext_vector_type(16))) _Float16 v16h;
typedef __attribute__((ext_vector_type(8)))  _Float16 v8h;
typedef __attribute__((ext_vector_type(4)))  _Float16 v4h;
typedef __attribute__((ext_vector_type(8)))  float    v8f;

// ---------------- WMMA helpers (CDNA5 wave32, 16x16x32 f16 -> f32) ----------

__device__ __forceinline__ v8f wmma16x16x32(v16h a, v16h b, v8f c) {
  return __builtin_amdgcn_wmma_f32_16x16x32_f16(false, a, false, b, (short)0, c,
                                                false, false);
}
__device__ __forceinline__ float lrelu02(float x){ return x >= 0.f ? x : 0.2f*x; }
__device__ __forceinline__ float finite0(float v){
  return ((__float_as_uint(v) & 0x7F800000u) != 0x7F800000u) ? v : 0.f;
}
__device__ __forceinline__ void atomicMaxF(float* a, float v){
  if (!(v == v)) return;
  if (v >= 0.f) atomicMax((int*)a, __float_as_int(v));
  else          atomicMin((unsigned int*)a, __float_as_uint(v));
}
__device__ __forceinline__ void waitlds(){
  asm volatile("s_wait_dscnt 0" ::: "memory");   // wave-local LDS ordering (CDNA5)
}

// A-fragment: 16 x KP f16 tile in LDS, row-major. Lane L<16 -> row L, K = {kb..kb+7, kb+16..kb+23}
__device__ __forceinline__ v16h frag_a(const _Float16* tile, int KP, int lane, int kt){
  const int m  = lane & 15;
  const int kb = kt*32 + ((lane >> 4) & 1)*8;
  const _Float16* p = tile + m*KP + kb;
  v8h lo = *(const v8h*)(p);
  v8h hi = *(const v8h*)(p + 16);
  v16h a;
#pragma unroll
  for (int j = 0; j < 8; ++j){ a[j] = lo[j]; a[j+8] = hi[j]; }
  return a;
}
// B-fragment: weights pre-swizzled into fragment order -> one contiguous 32B load per lane
__device__ __forceinline__ v16h frag_b(const _Float16* wsw, int NT, int lane, int kt, int nt){
  return *(const v16h*)(wsw + (size_t)(((kt*NT + nt)*32) + lane)*16);
}

// Feature-row permutations (LDS feature index k -> reference weight row)
struct PermId { __device__ int operator()(int k) const { return k; } };
struct PermQ  { __device__ int operator()(int k) const { return k < 64 ? k + 10 : k - 64; } };
struct PermM  { __device__ int operator()(int k) const {
                  return k < 64 ? k + 10 : (k < 74 ? k - 64 : k); } };
struct PermR  { __device__ int operator()(int k) const { return k < 128 ? k + 10 : k - 128; } };

// Convert f32 weights [in x out] (rows addressed through perm) to f16 fragment-order LDS,
// zero-padded to KP x NP. in_valid counts valid LDS-space feature rows.
template<typename P>
__device__ void load_w_sw(_Float16* dst, const float* src, int in_valid, int out_d,
                          int KP, int NP, int tid, int nth, P perm){
  const int NT = NP >> 4;
  const int total = KP * NP;
  for (int i = tid; i < total; i += nth) {
    int j    = i & 15;
    int lane = (i >> 4) & 31;
    int tile = i >> 9;
    int nt = tile % NT;
    int kt = tile / NT;
    int k = kt*32 + ((lane >> 4) & 1)*8 + ((j < 8) ? j : (j + 8));
    int n = nt*16 + (lane & 15);
    float v = (k < in_valid && n < out_d) ? src[perm(k)*out_d + n] : 0.f;
    dst[i] = (_Float16)v;
  }
}

// Vectorized gather staging: copy nchunks*4 floats (16B-aligned src) into f16 LDS row,
// two lanes per row (h = 0/1) interleaving by float4 chunk.
__device__ __forceinline__ void stage_f4(_Float16* dst_row, const float* src_row,
                                         int nchunks, int h){
  for (int c = h; c < nchunks; c += 2){
    const float4 f = *(const float4*)(src_row + c*4);
    v4h o; o[0] = (_Float16)f.x; o[1] = (_Float16)f.y;
           o[2] = (_Float16)f.z; o[3] = (_Float16)f.w;
    *(v4h*)(dst_row + c*4) = o;
  }
}
__device__ __forceinline__ void stage_f4_fin(_Float16* dst_row, const float* src_row,
                                             int nchunks, int h){
  for (int c = h; c < nchunks; c += 2){
    const float4 f = *(const float4*)(src_row + c*4);
    v4h o; o[0] = (_Float16)finite0(f.x); o[1] = (_Float16)finite0(f.y);
           o[2] = (_Float16)finite0(f.z); o[3] = (_Float16)finite0(f.w);
    *(v4h*)(dst_row + c*4) = o;
  }
}

// 2-layer MLP on one 16-row tile: [16 x KP1] @ W1[KP1 x 64] + b1, lrelu, @ W2[64 x NP2].
// Raw (bias-free) layer-2 accumulators returned in C[NP2/16][8].
template<int KP1, int NP2>
__device__ __forceinline__ void mlp2(const _Float16* act1, _Float16* act2,
                                     const _Float16* W1, const float* B1,
                                     const _Float16* W2,
                                     float (&C)[NP2/16][8], int lane)
{
#pragma unroll
  for (int nt = 0; nt < 4; ++nt){
    v8f c = {};
#pragma unroll
    for (int kt = 0; kt < KP1/32; ++kt)
      c = wmma16x16x32(frag_a(act1, KP1, lane, kt), frag_b(W1, 4, lane, kt, nt), c);
    const int mH = ((lane >> 4) & 1)*8;
    const int n  = nt*16 + (lane & 15);
    const float bn = B1[n];
#pragma unroll
    for (int v = 0; v < 8; ++v)
      act2[(mH + v)*64 + n] = (_Float16)lrelu02(c[v] + bn);
  }
  waitlds();
#pragma unroll
  for (int nt = 0; nt < NP2/16; ++nt){
    v8f c = {};
#pragma unroll
    for (int kt = 0; kt < 2; ++kt)
      c = wmma16x16x32(frag_a(act2, 64, lane, kt), frag_b(W2, NP2/16, lane, kt, nt), c);
#pragma unroll
    for (int v = 0; v < 8; ++v) C[nt][v] = c[v];
  }
}

__device__ __forceinline__ void softmax7(const float* in, float* out){
  float m = in[0];
#pragma unroll
  for (int i = 1; i < 7; ++i) m = fmaxf(m, in[i]);
  float s = 0.f;
#pragma unroll
  for (int i = 0; i < 7; ++i){ out[i] = __expf(in[i] - m); s += out[i]; }
  float inv = 1.f / s;
#pragma unroll
  for (int i = 0; i < 7; ++i) out[i] *= inv;
}

// ---------------------------------- kernels ---------------------------------

__global__ void k_fill(float* __restrict__ p, long n, float v){
  long i  = (long)blockIdx.x*blockDim.x + threadIdx.x;
  long st = (long)gridDim.x*blockDim.x;
  for (; i < n; i += st) p[i] = v;
}

// x = lrelu(nf0 @ W + b)   [N x 10] -> [N x 64]
__global__ __launch_bounds__(128)
void k_dim_up(const float* __restrict__ nf0, const float* __restrict__ W,
              const float* __restrict__ B, float* __restrict__ x)
{
  __shared__ __align__(32) _Float16 sW[32*64];
  __shared__ __align__(32) _Float16 sA[4][16*32];
  const int tid = threadIdx.x, lane = tid & 31, w = tid >> 5;
  load_w_sw(sW, W, 10, 64, 32, 64, tid, 128, PermId{});
  __syncthreads();
  const long base = ((long)blockIdx.x*4 + w)*16;
  {
    const int r = lane & 15, h = lane >> 4;
    const long node = base + r;
    for (int k = h; k < 32; k += 2)
      sA[w][r*32 + k] = (_Float16)((k < 10) ? nf0[node*10 + k] : 0.f);
  }
  waitlds();
  const int mH = ((lane >> 4) & 1)*8, nl = lane & 15;
#pragma unroll
  for (int nt = 0; nt < 4; ++nt){
    v8f c = {};
    c = wmma16x16x32(frag_a(sA[w], 32, lane, 0), frag_b(sW, 4, lane, 0, nt), c);
    const int n = nt*16 + nl;
    const float bn = B[n];
#pragma unroll
    for (int v = 0; v < 8; ++v)
      x[(base + mH + v)*64 + n] = lrelu02(c[v] + bn);
  }
}

// x_i = MLP([h_src,h_dst,ef]); agg_i[dst] += x_i
__global__ __launch_bounds__(128)
void k_net_o2i(const float* __restrict__ x, const int* __restrict__ esrc,
               const int* __restrict__ edst, const float* __restrict__ ef,
               const float* __restrict__ W1, const float* __restrict__ B1,
               const float* __restrict__ W2, const float* __restrict__ B2,
               float* __restrict__ agg)
{
  constexpr int KP1 = 160, NP2 = 64;
  __shared__ __align__(32) _Float16 sW1[KP1*64];
  __shared__ __align__(32) _Float16 sW2[64*NP2];
  __shared__ __align__(32) _Float16 sA1[4][16*KP1];
  __shared__ __align__(32) _Float16 sA2[4][16*64];
  __shared__ int sDst[4][16];
  const int tid = threadIdx.x, lane = tid & 31, w = tid >> 5;
  load_w_sw(sW1, W1, 138, 64, KP1, 64, tid, 128, PermId{});
  load_w_sw(sW2, W2, 64, 64, 64, NP2, tid, 128, PermId{});
  __syncthreads();
  const long base = ((long)blockIdx.x*4 + w)*16;
  {
    const int r = lane & 15, h = lane >> 4;
    const long e = base + r;
    const int s = esrc[e], d = edst[e];
    if (h == 0) sDst[w][r] = d;
    _Float16* row = sA1[w] + r*KP1;
    stage_f4(row,      x + (long)s*64, 16, h);
    stage_f4(row + 64, x + (long)d*64, 16, h);
    for (int k = 128 + h; k < KP1; k += 2)
      row[k] = (_Float16)((k < 138) ? ef[e*10 + (k - 128)] : 0.f);
  }
  waitlds();
  float C[NP2/16][8];
  mlp2<KP1, NP2>(sA1[w], sA2[w], sW1, B1, sW2, C, lane);
  const int mH = ((lane >> 4) & 1)*8;
#pragma unroll
  for (int nt = 0; nt < 4; ++nt){
    const int n = nt*16 + (lane & 15);
    const float bn = B2[n];
#pragma unroll
    for (int v = 0; v < 8; ++v)
      atomicAdd(&agg[(long)sDst[w][mH + v]*64 + n], C[nt][v] + bn);
  }
}

// x_o = MLP([h_dst,h_src,ef]); k=sig(x_o[0]); nfo1[src]+=k*f1; nfo2[src]=max(.,k*f2)
__global__ __launch_bounds__(128)
void k_net_i2o(const float* __restrict__ x, const int* __restrict__ esrc,
               const int* __restrict__ edst, const float* __restrict__ ef,
               const float* __restrict__ W1, const float* __restrict__ B1,
               const float* __restrict__ W2, const float* __restrict__ B2,
               float* __restrict__ nfo1, float* __restrict__ nfo2)
{
  constexpr int KP1 = 160, NP2 = 80;   // 65 -> pad 80
  __shared__ __align__(32) _Float16 sW1[KP1*64];
  __shared__ __align__(32) _Float16 sW2[64*NP2];
  __shared__ __align__(32) _Float16 sA1[4][16*KP1];
  __shared__ __align__(32) _Float16 sA2[4][16*64];
  __shared__ int sSrc[4][16];
  const int tid = threadIdx.x, lane = tid & 31, w = tid >> 5;
  load_w_sw(sW1, W1, 138, 64, KP1, 64, tid, 128, PermId{});
  load_w_sw(sW2, W2, 64, 65, 64, NP2, tid, 128, PermId{});
  __syncthreads();
  const long base = ((long)blockIdx.x*4 + w)*16;
  {
    const int r = lane & 15, h = lane >> 4;
    const long e = base + r;
    const int s = esrc[e], d = edst[e];
    if (h == 0) sSrc[w][r] = s;
    _Float16* row = sA1[w] + r*KP1;
    stage_f4(row,      x + (long)d*64, 16, h);   // h_dst first
    stage_f4(row + 64, x + (long)s*64, 16, h);   // then h_src
    for (int k = 128 + h; k < KP1; k += 2)
      row[k] = (_Float16)((k < 138) ? ef[e*10 + (k - 128)] : 0.f);
  }
  waitlds();
  float C[5][8];
  mlp2<KP1, NP2>(sA1[w], sA2[w], sW1, B1, sW2, C, lane);
  const int h = lane >> 4, mH = h*8;
  float kg[8];
  {
    const float b0 = B2[0];
#pragma unroll
    for (int v = 0; v < 8; ++v){
      float g = __shfl(C[0][v], h*16, 32);       // column-0 holder = lane h*16
      kg[v] = 1.f/(1.f + __expf(-(g + b0)));
    }
  }
#pragma unroll
  for (int nt = 0; nt < 5; ++nt){
    const int n = nt*16 + (lane & 15);
    const float bn = (n < 65) ? B2[n] : 0.f;
    if (n < 1 || n > 64) continue;               // lane-uniform over v
#pragma unroll
    for (int v = 0; v < 8; ++v){
      const float val = (C[nt][v] + bn)*kg[v];
      const long s = sSrc[w][mH + v];
      if (n <= 32) atomicAdd(&nfo1[s*32 + (n - 1)], val);
      else         atomicMaxF(&nfo2[s*32 + (n - 33)], val);
    }
  }
}

// out_o = MLP([x,nfo1,nfo2]); x = (mask? out_o : agg_i) + x
__global__ __launch_bounds__(128)
void k_nc_reduce(float* x, const float* __restrict__ nfo1,
                 const float* __restrict__ nfo2, const float* __restrict__ aggi,
                 const unsigned char* __restrict__ mask,
                 const float* __restrict__ W1, const float* __restrict__ B1,
                 const float* __restrict__ W2, const float* __restrict__ B2)
{
  constexpr int KP1 = 128, NP2 = 64;
  __shared__ __align__(32) _Float16 sW1[KP1*64];
  __shared__ __align__(32) _Float16 sW2[64*NP2];
  __shared__ __align__(32) _Float16 sA1[4][16*KP1];
  __shared__ __align__(32) _Float16 sA2[4][16*64];
  const int tid = threadIdx.x, lane = tid & 31, w = tid >> 5;
  load_w_sw(sW1, W1, 128, 64, KP1, 64, tid, 128, PermId{});
  load_w_sw(sW2, W2, 64, 64, 64, NP2, tid, 128, PermId{});
  __syncthreads();
  const long base = ((long)blockIdx.x*4 + w)*16;
  {
    const int r = lane & 15, h = lane >> 4;
    const long node = base + r;
    _Float16* row = sA1[w] + r*KP1;
    stage_f4    (row,      x    + node*64, 16, h);
    stage_f4    (row + 64, nfo1 + node*32,  8, h);
    stage_f4_fin(row + 96, nfo2 + node*32,  8, h);
  }
  waitlds();
  float C[4][8];
  mlp2<KP1, NP2>(sA1[w], sA2[w], sW1, B1, sW2, C, lane);
  const int mH = ((lane >> 4) & 1)*8;
#pragma unroll
  for (int nt = 0; nt < 4; ++nt){
    const int n = nt*16 + (lane & 15);
    const float bn = B2[n];
#pragma unroll
    for (int v = 0; v < 8; ++v){
      const long node = base + mH + v;
      const float o   = C[nt][v] + bn;
      const float sel = mask[node] ? o : aggi[node*64 + n];
      x[node*64 + n]  = sel + x[node*64 + n];
    }
  }
}

// net_delays = MLP(x) : 64 -> 64 -> 4
__global__ __launch_bounds__(128)
void k_netdelay(const float* __restrict__ x,
                const float* __restrict__ W1, const float* __restrict__ B1,
                const float* __restrict__ W2, const float* __restrict__ B2,
                float* __restrict__ out)
{
  constexpr int KP1 = 64, NP2 = 16;
  __shared__ __align__(32) _Float16 sW1[KP1*64];
  __shared__ __align__(32) _Float16 sW2[64*NP2];
  __shared__ __align__(32) _Float16 sA1[4][16*KP1];
  __shared__ __align__(32) _Float16 sA2[4][16*64];
  const int tid = threadIdx.x, lane = tid & 31, w = tid >> 5;
  load_w_sw(sW1, W1, 64, 64, KP1, 64, tid, 128, PermId{});
  load_w_sw(sW2, W2, 64, 4, 64, NP2, tid, 128, PermId{});
  __syncthreads();
  const long base = ((long)blockIdx.x*4 + w)*16;
  {
    const int r = lane & 15, h = lane >> 4;
    stage_f4(sA1[w] + r*KP1, x + (base + r)*64, 16, h);
  }
  waitlds();
  float C[1][8];
  mlp2<KP1, NP2>(sA1[w], sA2[w], sW1, B1, sW2, C, lane);
  const int mH = ((lane >> 4) & 1)*8, n = lane & 15;
  if (n < 4){
    const float bn = B2[n];
#pragma unroll
    for (int v = 0; v < 8; ++v)
      out[(base + mH + v)*4 + n] = C[0][v] + bn;
  }
}

// q = softmax(MLP(nf1[dst]).reshape(8,2,7)); lutv = q0 . LUT . q1
__global__ __launch_bounds__(128)
void k_prop_q(const float* __restrict__ nf0, const float* __restrict__ x,
              const int* __restrict__ cdst, const float* __restrict__ lut,
              const float* __restrict__ W1, const float* __restrict__ B1,
              const float* __restrict__ W2, const float* __restrict__ B2,
              float* __restrict__ lutv)
{
  constexpr int KP1 = 96, NP2 = 112;
  __shared__ __align__(32) _Float16 sW1[KP1*64];
  __shared__ __align__(32) _Float16 sW2[64*NP2];
  __shared__ __align__(32) _Float16 sA2[4][16*64];
  __shared__ __align__(32) char uni[4*16*NP2*4];   // per-wave: act1 (f16) then q (f32)
  const int tid = threadIdx.x, lane = tid & 31, w = tid >> 5;
  _Float16* a1 = (_Float16*)(uni + (size_t)w*(16*NP2*4));
  float*    qb = (float*)   (uni + (size_t)w*(16*NP2*4));
  load_w_sw(sW1, W1, 74, 64, KP1, 64, tid, 128, PermQ{});   // LDS order: [x, nf0]
  load_w_sw(sW2, W2, 64, 112, 64, NP2, tid, 128, PermId{});
  __syncthreads();
  const long base = ((long)blockIdx.x*4 + w)*16;
  {
    const int r = lane & 15, h = lane >> 4;
    const long e = base + r;
    const int d = cdst[e];
    _Float16* row = a1 + r*KP1;
    stage_f4(row, x + (long)d*64, 16, h);
    for (int k = 64 + h; k < KP1; k += 2)
      row[k] = (_Float16)((k < 74) ? nf0[(long)d*10 + (k - 64)] : 0.f);
  }
  waitlds();
  float C[7][8];
  mlp2<KP1, NP2>(a1, sA2[w], sW1, B1, sW2, C, lane);
  {
    const int mH = ((lane >> 4) & 1)*8, nl = lane & 15;
#pragma unroll
    for (int nt = 0; nt < 7; ++nt){
      const int n = nt*16 + nl;
      const float bn = B2[n];
#pragma unroll
      for (int v = 0; v < 8; ++v) qb[(mH + v)*NP2 + n] = C[nt][v] + bn;
    }
  }
  waitlds();
  {
    const int r = lane & 15, h = lane >> 4;
    const long e = base + r;
    const float* qr = qb + r*NP2;
#pragma unroll
    for (int li = 0; li < 4; ++li){
      const int l = h*4 + li;
      float q0[7], q1[7];
      softmax7(qr + l*14,     q0);
      softmax7(qr + l*14 + 7, q1);
      const float* L = lut + ((long)e*8 + l)*49;
      float acc = 0.f;
#pragma unroll
      for (int rr = 0; rr < 7; ++rr){
        float dd = 0.f;
#pragma unroll
        for (int cc = 0; cc < 7; ++cc) dd += L[rr*7 + cc]*q1[cc];
        acc += q0[rr]*dd;
      }
      lutv[e*8 + l] = acc;
    }
  }
}

// m = MLP([nf1[src], lut_val]); k=sig(m0); agg[dst]+=k*m[1:65]; cell_delays=m[65:69]
__global__ __launch_bounds__(128)
void k_prop_msg(const float* __restrict__ nf0, const float* __restrict__ x,
                const int* __restrict__ csrc, const int* __restrict__ cdst,
                const float* __restrict__ lutv,
                const float* __restrict__ W1, const float* __restrict__ B1,
                const float* __restrict__ W2, const float* __restrict__ B2,
                float* __restrict__ agg, float* __restrict__ cd_out)
{
  constexpr int KP1 = 96, NP2 = 80;   // 82 -> 96, 69 -> 80
  __shared__ __align__(32) _Float16 sW1[KP1*64];
  __shared__ __align__(32) _Float16 sW2[64*NP2];
  __shared__ __align__(32) _Float16 sA1[4][16*KP1];
  __shared__ __align__(32) _Float16 sA2[4][16*64];
  __shared__ int sDst[4][16];
  const int tid = threadIdx.x, lane = tid & 31, w = tid >> 5;
  load_w_sw(sW1, W1, 82, 64, KP1, 64, tid, 128, PermM{});   // LDS order: [x, nf0, lutv]
  load_w_sw(sW2, W2, 64, 69, 64, NP2, tid, 128, PermId{});
  __syncthreads();
  const long base = ((long)blockIdx.x*4 + w)*16;
  {
    const int r = lane & 15, h = lane >> 4;
    const long e = base + r;
    const int s = csrc[e], d = cdst[e];
    if (h == 0) sDst[w][r] = d;
    _Float16* row = sA1[w] + r*KP1;
    stage_f4(row, x + (long)s*64, 16, h);
    for (int k = 64 + h; k < KP1; k += 2){
      float v = 0.f;
      if (k < 74)      v = nf0[(long)s*10 + (k - 64)];
      else if (k < 82) v = lutv[e*8 + (k - 74)];
      row[k] = (_Float16)v;
    }
  }
  waitlds();
  float C[5][8];
  mlp2<KP1, NP2>(sA1[w], sA2[w], sW1, B1, sW2, C, lane);
  const int h = lane >> 4, mH = h*8;
  float kg[8];
  {
    const float b0 = B2[0];
#pragma unroll
    for (int v = 0; v < 8; ++v){
      float g = __shfl(C[0][v], h*16, 32);
      kg[v] = 1.f/(1.f + __expf(-(g + b0)));
    }
  }
#pragma unroll
  for (int nt = 0; nt < 5; ++nt){
    const int n = nt*16 + (lane & 15);
    const float bn = (n < 69) ? B2[n] : 0.f;
#pragma unroll
    for (int v = 0; v < 8; ++v){
      const long e = base + mH + v;
      if (n >= 1 && n <= 64){
        atomicAdd(&agg[(long)sDst[w][mH + v]*64 + (n - 1)], (C[nt][v] + bn)*kg[v]);
      } else if (n >= 65 && n <= 68){
        cd_out[e*4 + (n - 65)] = C[nt][v] + bn;   // cell_delays: not gated
      }
    }
  }
}

// atslew = MLP([nf0, x, agg]) : 138 -> 64 -> 8
__global__ __launch_bounds__(128)
void k_prop_reduce(const float* __restrict__ nf0, const float* __restrict__ x,
                   const float* __restrict__ agg,
                   const float* __restrict__ W1, const float* __restrict__ B1,
                   const float* __restrict__ W2, const float* __restrict__ B2,
                   float* __restrict__ out)
{
  constexpr int KP1 = 160, NP2 = 16;   // 138 -> 160, 8 -> 16
  __shared__ __align__(32) _Float16 sW1[KP1*64];
  __shared__ __align__(32) _Float16 sW2[64*NP2];
  __shared__ __align__(32) _Float16 sA1[4][16*KP1];
  __shared__ __align__(32) _Float16 sA2[4][16*64];
  const int tid = threadIdx.x, lane = tid & 31, w = tid >> 5;
  load_w_sw(sW1, W1, 138, 64, KP1, 64, tid, 128, PermR{});  // LDS order: [x, agg, nf0]
  load_w_sw(sW2, W2, 64, 8, 64, NP2, tid, 128, PermId{});
  __syncthreads();
  const long base = ((long)blockIdx.x*4 + w)*16;
  {
    const int r = lane & 15, h = lane >> 4;
    const long node = base + r;
    _Float16* row = sA1[w] + r*KP1;
    stage_f4(row,      x   + node*64, 16, h);
    stage_f4(row + 64, agg + node*64, 16, h);
    for (int k = 128 + h; k < KP1; k += 2)
      row[k] = (_Float16)((k < 138) ? nf0[node*10 + (k - 128)] : 0.f);
  }
  waitlds();
  float C[1][8];
  mlp2<KP1, NP2>(sA1[w], sA2[w], sW1, B1, sW2, C, lane);
  const int mH = ((lane >> 4) & 1)*8, n = lane & 15;
  if (n < 8){
    const float bn = B2[n];
#pragma unroll
    for (int v = 0; v < 8; ++v)
      out[(base + mH + v)*8 + n] = C[0][v] + bn;
  }
}

// --------------------------------- launcher ---------------------------------
// param leaves (jax pytree, dict keys sorted):
//  8: dim_up W, 9: dim_up b
//  10+12l .. : nc[l] { i2o: W1,b1,W2,b2 ; o2i: W1,b1,W2,b2 ; reduce: W1,b1,W2,b2 }
//  58..61: prop_msg ; 62..65: prop_q ; 66..69: prop_reduce ; 70..73: to_netdelay

extern "C" void kernel_launch(void* const* d_in, const int* in_sizes, int n_in,
                              void* d_out, int out_size, void* d_ws, size_t ws_size,
                              hipStream_t stream)
{
  (void)in_sizes; (void)n_in; (void)out_size; (void)ws_size;
  const float* nf0      = (const float*)d_in[0];
  const int*   net_src  = (const int*)d_in[1];
  const int*   net_dst  = (const int*)d_in[2];
  const float* net_ef   = (const float*)d_in[3];
  const int*   cell_src = (const int*)d_in[4];
  const int*   cell_dst = (const int*)d_in[5];
  const float* cell_lut = (const float*)d_in[6];
  const unsigned char* omask = (const unsigned char*)d_in[7];
  auto P = [&](int i){ return (const float*)d_in[i]; };

  float* xbuf = (float*)d_ws;
  float* aggA = xbuf + (size_t)N_NODES*64;
  float* nfo1 = aggA + (size_t)N_NODES*64;
  float* nfo2 = nfo1 + (size_t)N_NODES*32;
  float* lutv = nfo2 + (size_t)N_NODES*32;

  float* out_nd = (float*)d_out;                       // N x 4
  float* out_cd = out_nd + (size_t)N_NODES*4;          // E_CELL x 4
  float* out_as = out_cd + (size_t)E_CELL_N*4;         // N x 8

  const int gridN = N_NODES  / 64;
  const int gridE = E_NET_N  / 64;
  const int gridC = E_CELL_N / 64;

  k_dim_up<<<gridN, 128, 0, stream>>>(nf0, P(8), P(9), xbuf);

  for (int l = 0; l < 4; ++l){
    const int b = 10 + 12*l;
    k_fill<<<1024, 256, 0, stream>>>(aggA, (long)N_NODES*64, 0.f);
    k_fill<<<1024, 256, 0, stream>>>(nfo1, (long)N_NODES*32, 0.f);
    k_fill<<<1024, 256, 0, stream>>>(nfo2, (long)N_NODES*32, -INFINITY);
    k_net_o2i<<<gridE, 128, 0, stream>>>(xbuf, net_src, net_dst, net_ef,
                P(b+4), P(b+5), P(b+6), P(b+7), aggA);
    k_net_i2o<<<gridE, 128, 0, stream>>>(xbuf, net_src, net_dst, net_ef,
                P(b), P(b+1), P(b+2), P(b+3), nfo1, nfo2);
    k_nc_reduce<<<gridN, 128, 0, stream>>>(xbuf, nfo1, nfo2, aggA, omask,
                P(b+8), P(b+9), P(b+10), P(b+11));
  }

  k_netdelay<<<gridN, 128, 0, stream>>>(xbuf, P(70), P(71), P(72), P(73), out_nd);

  k_prop_q<<<gridC, 128, 0, stream>>>(nf0, xbuf, cell_dst, cell_lut,
              P(62), P(63), P(64), P(65), lutv);

  k_fill<<<1024, 256, 0, stream>>>(aggA, (long)N_NODES*64, 0.f);
  k_prop_msg<<<gridC, 128, 0, stream>>>(nf0, xbuf, cell_src, cell_dst, lutv,
              P(58), P(59), P(60), P(61), aggA, out_cd);

  k_prop_reduce<<<gridN, 128, 0, stream>>>(nf0, xbuf, aggA,
              P(66), P(67), P(68), P(69), out_as);
}
